// TransformAttention_1348619731019
// MI455X (gfx1250) — matrix operand order repaired and checked by reading, not compiled, not run on hardware
//
#include <hip/hip_runtime.h>

// ---------------- problem constants ----------------
#define BB   16
#define LL   24
#define NN   325
#define DH   256
#define HH   8
#define CC   32          // DH / HH chunks
#define CB   (CC * BB)   // 512
#define TOK  (BB * LL * NN)          // 124800 tokens
#define INVS 0.17677669529663687f    // 1 / sqrt(32)
#define EPSV 1e-5f

#define AS1 __attribute__((address_space(1)))
#define AS3 __attribute__((address_space(3)))

typedef __attribute__((ext_vector_type(16))) __bf16 v16bf16;
typedef __attribute__((ext_vector_type(8)))  float  v8f;
typedef __attribute__((ext_vector_type(4)))  int    i4;
typedef __attribute__((ext_vector_type(4)))  float  f4;

union AFrag { unsigned u[8]; i4 q[2]; v16bf16 v; };

// fp32 -> bf16 round-to-nearest-even (scalar; used in tiny prep kernel)
__device__ __forceinline__ unsigned short f2bf(float f) {
    unsigned u = __builtin_bit_cast(unsigned, f);
    u += 0x7FFFu + ((u >> 16) & 1u);
    return (unsigned short)(u >> 16);
}

// pack two fp32 -> dword of two bf16, round-half-up: 2x add + 1x v_perm_b32
__device__ __forceinline__ unsigned packbf2(float a, float b) {
    unsigned ua = __builtin_bit_cast(unsigned, a) + 0x8000u;
    unsigned ub = __builtin_bit_cast(unsigned, b) + 0x8000u;
#if __has_builtin(__builtin_amdgcn_perm)
    return __builtin_amdgcn_perm(ub, ua, 0x07060302u);  // {ua.hi16, ub.hi16}
#else
    return (ua >> 16) | (ub & 0xffff0000u);
#endif
}

// 16-byte global -> LDS copy; async (ASYNCcnt) when the builtin exists
__device__ __forceinline__ void cp16(void* lds, const void* g) {
#if __has_builtin(__builtin_amdgcn_global_load_async_to_lds_b128)
    __builtin_amdgcn_global_load_async_to_lds_b128(
        (AS1 i4*)g, (AS3 i4*)lds, 0, 0);
#else
    *(i4*)lds = *(const i4*)g;
#endif
}

__device__ __forceinline__ void wait_async0() {
#if __has_builtin(__builtin_amdgcn_global_load_async_to_lds_b128)
#if __has_builtin(__builtin_amdgcn_s_wait_asynccnt)
    __builtin_amdgcn_s_wait_asynccnt(0);
#else
    asm volatile("s_wait_asynccnt 0x0" ::: "memory");
#endif
#endif
}

// ---------------- zero the stats accumulators ----------------
__global__ void init_stats(float* stats) {
    for (int i = threadIdx.x; i < 2048; i += 256) stats[i] = 0.f;
}

// ---------------- one-time W fp32[k][n] -> bf16 Wt[n][k] ----------------
__global__ __launch_bounds__(256)
void prep_w(const float* __restrict__ W, unsigned short* __restrict__ Wt) {
    int idx = blockIdx.x * 256 + threadIdx.x;     // 65536 elements
    int n = idx >> 8, k = idx & 255;
    Wt[(size_t)n * DH + k] = f2bf(W[(size_t)k * DH + n]);
}

// ---------------- Y = X @ W + b, plus per-channel sum/sumsq -------------
// grid = TOK/128 blocks of 256 threads (8 waves, 2x4 wave grid).
// Block tile: 128(M) x 256(N=all channels); wave tile 64x64 = 16 WMMAs/k-step.
// X: global->VGPR->bf16 convert->LDS (converted once per element globally).
// W: bf16, async global->LDS (ASYNCcnt). Double-buffered stages.
struct XR { f4 v[4]; };

__global__ __launch_bounds__(256)
void gemm_bn(const float* __restrict__ X, const unsigned short* __restrict__ Wt,
             const float* __restrict__ bias, float* __restrict__ Y,
             float* __restrict__ statSum, float* __restrict__ statSsq)
{
    __shared__ unsigned short sXb[2][128][40];  // bf16 [m][k], 80B row stride
    __shared__ unsigned short sWb[2][256][40];  // bf16 [n][k], 80B row stride
    __shared__ float sSum[256], sSsq[256];

    const int m0   = blockIdx.x * 128;
    const int tid  = threadIdx.x;
    const int lane = tid & 31;
    const int wave = tid >> 5;
    const int wy   = wave >> 2;        // 0..1 : M offset wy*64
    const int wx   = wave & 3;         // 0..3 : N offset wx*64
    const int hi   = (lane >> 4) & 1;
    const int l15  = lane & 15;

    sSum[tid] = 0.f; sSsq[tid] = 0.f;

    // ---- stage helpers ----
    auto loadX = [&](int kiter) {                 // 16 fp32 per thread
        XR x;
        const int r = tid >> 1, half = tid & 1;
        const float* src = &X[(size_t)(m0 + r) * DH + kiter * 32 + half * 16];
#pragma unroll
        for (int j = 0; j < 4; ++j) x.v[j] = *(const f4*)(src + j * 4);
        return x;
    };
    auto storeX = [&](const XR& x, int bb) {      // convert + 2x ds_store_b128
        const int r = tid >> 1, half = tid & 1;
        unsigned pk[8];
#pragma unroll
        for (int j = 0; j < 8; ++j)
            pk[j] = packbf2(x.v[j >> 1][(j & 1) * 2], x.v[j >> 1][(j & 1) * 2 + 1]);
        *(i4*)&sXb[bb][r][half * 16]     = *(i4*)&pk[0];
        *(i4*)&sXb[bb][r][half * 16 + 8] = *(i4*)&pk[4];
    };
    auto issueW = [&](int kiter, int bb) {        // 4 async 16B copies / thread
#pragma unroll
        for (int cg = 0; cg < 4; ++cg)
            cp16(&sWb[bb][tid][cg * 8],
                 &Wt[(size_t)tid * DH + kiter * 32 + cg * 8]);
    };

    // ---- prologue: stage 0 ----
    {
        XR x0 = loadX(0);
        issueW(0, 0);
        storeX(x0, 0);
        wait_async0();
    }
    __syncthreads();

    v8f acc[4][4] = {};

    for (int k = 0; k < 8; ++k) {
        const int bb = k & 1;
        XR xn;
        if (k < 7) { xn = loadX(k + 1); issueW(k + 1, bb ^ 1); }

        // ---- compute: 4 B-frags cached, stream 4 A-frags, 16 WMMAs ----
        AFrag bfr[4];
#pragma unroll
        for (int nb = 0; nb < 4; ++nb) {
            const int brow = wx * 64 + nb * 16 + l15;
            bfr[nb].q[0] = *(const i4*)&sWb[bb][brow][hi * 16];
            bfr[nb].q[1] = *(const i4*)&sWb[bb][brow][hi * 16 + 8];
        }
#pragma unroll
        for (int mb = 0; mb < 4; ++mb) {
            AFrag a;
            const int arow = wy * 64 + mb * 16 + l15;
            a.q[0] = *(const i4*)&sXb[bb][arow][hi * 16];
            a.q[1] = *(const i4*)&sXb[bb][arow][hi * 16 + 8];
#pragma unroll
            for (int nb = 0; nb < 4; ++nb)
                acc[mb][nb] = __builtin_amdgcn_wmma_f32_16x16x32_bf16(
                    false, a.v, false, bfr[nb].v, (short)0, acc[mb][nb],
                    false, false);
        }

        if (k < 7) { storeX(xn, bb ^ 1); wait_async0(); }
        __syncthreads();
    }

    // ---- epilogue: bias, store, per-channel stats ----
#pragma unroll
    for (int nb = 0; nb < 4; ++nb) {
        const int n  = wx * 64 + nb * 16 + l15;   // global channel (N==DH)
        const float bv = bias[n];
        float s1 = 0.f, s2 = 0.f;
#pragma unroll
        for (int mb = 0; mb < 4; ++mb) {
#pragma unroll
            for (int r = 0; r < 8; ++r) {
                float y = acc[mb][nb][r] + bv;
                int m = m0 + wy * 64 + mb * 16 + hi * 8 + r;
                Y[(size_t)m * DH + n] = y;
                s1 += y; s2 += y * y;
            }
        }
        atomicAdd(&sSum[n], s1);
        atomicAdd(&sSsq[n], s2);
    }
    __syncthreads();
    atomicAdd(&statSum[tid], sSum[tid]);
    atomicAdd(&statSsq[tid], sSsq[tid]);
}

// ---------------- sums -> per-channel scale/shift ----------------
__global__ void finalize_bn(const float* __restrict__ sum, const float* __restrict__ ssq,
                            const float* __restrict__ gamma, const float* __restrict__ beta,
                            float* __restrict__ scale, float* __restrict__ shift)
{
    int ch = threadIdx.x;                    // 256 threads
    float inv = 1.f / (float)TOK;
    float mu  = sum[ch] * inv;
    float var = ssq[ch] * inv - mu * mu;     // biased, matches jnp.var
    float sc  = gamma[ch] * rsqrtf(var + EPSV);
    scale[ch] = sc;
    shift[ch] = beta[ch] - mu * sc;
}

// ---------------- attention: one wave per (cb, n) group ----------------
// grid = (CB*NN)/8 blocks of 256 threads (8 waves); 512*325 = 8*20800 exactly
__global__ __launch_bounds__(256)
void attention(const float* __restrict__ Yq, const float* __restrict__ Yk,
               const float* __restrict__ Yv, const float* __restrict__ sc,
               float* __restrict__ Out)
{
    __shared__ float sK[8][LL][9];
    __shared__ float sV[8][LL][9];

    const int wave = threadIdx.x >> 5;
    const int lane = threadIdx.x & 31;
    const int grp  = blockIdx.x * 8 + wave;      // 0 .. CB*NN-1
    const int cb   = grp / NN;
    const int n    = grp % NN;
    const int cch  = cb >> 4;                    // chunk index (cb = c*B + b)
    const int b    = cb & 15;
    const int ch   = cch * HH;                   // channel base

    const float* scq = sc;         const float* shq = sc + 256;
    const float* sck = sc + 512;   const float* shk = sc + 768;
    const float* scv = sc + 1024;  const float* shv = sc + 1280;

    float q[HH];
    size_t base = 0;
    if (lane < LL) {
        const int l = lane;
        base = ((size_t)((b * LL + l) * NN + n)) * DH + ch;
#pragma unroll
        for (int h = 0; h < HH; ++h) {
            float qv = fmaxf(Yq[base + h] * scq[ch + h] + shq[ch + h], 0.f);
            q[h] = qv * INVS;                    // fold 1/sqrt(32) score scale
            sK[wave][l][h] = fmaxf(Yk[base + h] * sck[ch + h] + shk[ch + h], 0.f);
            sV[wave][l][h] = fmaxf(Yv[base + h] * scv[ch + h] + shv[ch + h], 0.f);
        }
    }
    __syncthreads();

    if (lane < LL) {
        float s[LL];
        float mx = -3.4e38f;
#pragma unroll
        for (int p = 0; p < LL; ++p) {
            float d = 0.f;
#pragma unroll
            for (int h = 0; h < HH; ++h) d += q[h] * sK[wave][p][h];
            s[p] = d;
            mx = fmaxf(mx, d);
        }
        float den = 0.f;
#pragma unroll
        for (int p = 0; p < LL; ++p) { float e = __expf(s[p] - mx); s[p] = e; den += e; }
        float rden = 1.f / den;
        float o[HH];
#pragma unroll
        for (int h = 0; h < HH; ++h) o[h] = 0.f;
#pragma unroll
        for (int p = 0; p < LL; ++p) {
            float w = s[p] * rden;
#pragma unroll
            for (int h = 0; h < HH; ++h) o[h] += w * sV[wave][p][h];
        }
#pragma unroll
        for (int h = 0; h < HH; ++h) Out[base + h] = o[h];
    }
}

// ---------------- final BN + ReLU elementwise ----------------
__global__ __launch_bounds__(256)
void bn_relu_out(const float* __restrict__ Yo, const float* __restrict__ scale,
                 const float* __restrict__ shift, float* __restrict__ out, size_t total)
{
    for (size_t i = (size_t)blockIdx.x * blockDim.x + threadIdx.x; i < total;
         i += (size_t)gridDim.x * blockDim.x) {
        int ch = (int)(i & (DH - 1));
        out[i] = fmaxf(Yo[i] * scale[ch] + shift[ch], 0.f);
    }
}

// ---------------- host launcher ----------------
extern "C" void kernel_launch(void* const* d_in, const int* in_sizes, int n_in,
                              void* d_out, int out_size, void* d_ws, size_t ws_size,
                              hipStream_t stream) {
    (void)in_sizes; (void)n_in; (void)out_size; (void)ws_size;
    const float* x  = (const float*)d_in[0];
    const float* sp = (const float*)d_in[1];   // ste_previous -> K
    const float* sf = (const float*)d_in[2];   // ste_future   -> Q
    const float* Wq = (const float*)d_in[3];  const float* bq = (const float*)d_in[4];
    const float* gq = (const float*)d_in[5];  const float* eq = (const float*)d_in[6];
    const float* Wk = (const float*)d_in[7];  const float* bk = (const float*)d_in[8];
    const float* gk = (const float*)d_in[9];  const float* ek = (const float*)d_in[10];
    const float* Wv = (const float*)d_in[11]; const float* bv = (const float*)d_in[12];
    const float* gv = (const float*)d_in[13]; const float* ev = (const float*)d_in[14];
    const float* Wo = (const float*)d_in[15]; const float* bo = (const float*)d_in[16];
    const float* go = (const float*)d_in[17]; const float* eo = (const float*)d_in[18];

    float* ws    = (float*)d_ws;
    float* stats = ws;                 // [4 sets][sum 256 | ssq 256]
    float* scsh  = ws + 2048;          // [4 sets][scale 256 | shift 256]
    unsigned short* wt = (unsigned short*)(ws + 4096);   // 4 x 256*256 bf16
    unsigned short* wtq = wt;
    unsigned short* wtk = wt + 65536;
    unsigned short* wtv = wt + 131072;
    unsigned short* wto = wt + 196608;
    const size_t BS = (size_t)TOK * DH;
    float* buf0 = ws + 4096 + 131072;  // Yq, later Yo
    float* buf1 = buf0 + BS;           // Yk
    float* buf2 = buf1 + BS;           // Yv
    float* buf3 = buf2 + BS;           // attention output (B,L,N,D)

    init_stats<<<1, 256, 0, stream>>>(stats);
    prep_w<<<256, 256, 0, stream>>>(Wq, wtq);
    prep_w<<<256, 256, 0, stream>>>(Wk, wtk);
    prep_w<<<256, 256, 0, stream>>>(Wv, wtv);
    prep_w<<<256, 256, 0, stream>>>(Wo, wto);

    const int ggrid = TOK / 128;   // 975 blocks
    gemm_bn<<<ggrid, 256, 0, stream>>>(sf, wtq, bq, buf0, stats + 0,    stats + 256);
    gemm_bn<<<ggrid, 256, 0, stream>>>(sp, wtk, bk, buf1, stats + 512,  stats + 768);
    gemm_bn<<<ggrid, 256, 0, stream>>>(x,  wtv, bv, buf2, stats + 1024, stats + 1280);

    finalize_bn<<<1, 256, 0, stream>>>(stats + 0,    stats + 256,  gq, eq, scsh + 0,    scsh + 256);
    finalize_bn<<<1, 256, 0, stream>>>(stats + 512,  stats + 768,  gk, ek, scsh + 512,  scsh + 768);
    finalize_bn<<<1, 256, 0, stream>>>(stats + 1024, stats + 1280, gv, ev, scsh + 1024, scsh + 1280);

    attention<<<(CB * NN) / 8, 256, 0, stream>>>(buf0, buf1, buf2, scsh, buf3);

    gemm_bn<<<ggrid, 256, 0, stream>>>(buf3, wto, bo, buf0, stats + 1536, stats + 1792);
    finalize_bn<<<1, 256, 0, stream>>>(stats + 1536, stats + 1792, go, eo, scsh + 1536, scsh + 1792);

    bn_relu_out<<<4096, 256, 0, stream>>>(buf0, scsh + 1536, scsh + 1792,
                                          (float*)d_out, BS);
}
